// MultiheadAttention_10866267259353
// MI455X (gfx1250) — compile-verified, matrix operand order checked by reading, not run on hardware
//
#include <hip/hip_runtime.h>

#define BATCH 2
#define SEQ   2048
#define DIMN  1024
#define HEADS 16
#define DHEAD 64
#define MTOT  (BATCH * SEQ)   // 4096 flattened (b, l)

typedef __bf16 bf16_t;
typedef __attribute__((ext_vector_type(16))) __bf16 v16bf;
typedef __attribute__((ext_vector_type(8)))  __bf16 v8bf;
typedef __attribute__((ext_vector_type(8)))  float  v8f;

static __device__ __forceinline__ bf16_t f2bf(float f) { return (bf16_t)f; }

// Load one 16x32 bf16 A-frag (or 32x16 B-frag, same pattern) per the CDNA5
// 16-bit WMMA VGPR layout: lane L holds row/col (L&15), K-chunks at
// hi*8..hi*8+7 and 16+hi*8..16+hi*8+7 where hi = L>>4.
static __device__ __forceinline__ v16bf load_frag(const bf16_t* base, int stride, int lane) {
  const int row = lane & 15;
  const int hi  = (lane >> 4) & 1;
  const bf16_t* rp = base + row * stride + hi * 8;
  v8bf c0 = *(const v8bf*)(rp);
  v8bf c1 = *(const v8bf*)(rp + 16);
  v16bf a;
#pragma unroll
  for (int i = 0; i < 8; ++i) { a[i] = c0[i]; a[i + 8] = c1[i]; }
  return a;
}

static __device__ __forceinline__ v8f wmma_bf16(v16bf a, v16bf b, v8f c) {
  return __builtin_amdgcn_wmma_f32_16x16x32_bf16(false, a, false, b, (short)0, c, false, false);
}

// Async global->LDS copies (ASYNCcnt). Instruction offset applies to both the
// global and LDS addresses per the CDNA5 async pseudocode.
static __device__ __forceinline__ void async_copy32(const bf16_t* g, bf16_t* l) {
  unsigned la = (unsigned)(size_t)l;                 // low 32 bits = LDS offset
  unsigned long long ga = (unsigned long long)(size_t)g;
  asm volatile("global_load_async_to_lds_b128 %0, %1, off"           :: "v"(la), "v"(ga) : "memory");
  asm volatile("global_load_async_to_lds_b128 %0, %1, off offset:16" :: "v"(la), "v"(ga) : "memory");
}
static __device__ __forceinline__ void async_copy64(const bf16_t* g, bf16_t* l) {
  unsigned la = (unsigned)(size_t)l;
  unsigned long long ga = (unsigned long long)(size_t)g;
  asm volatile("global_load_async_to_lds_b128 %0, %1, off"           :: "v"(la), "v"(ga) : "memory");
  asm volatile("global_load_async_to_lds_b128 %0, %1, off offset:16" :: "v"(la), "v"(ga) : "memory");
  asm volatile("global_load_async_to_lds_b128 %0, %1, off offset:32" :: "v"(la), "v"(ga) : "memory");
  asm volatile("global_load_async_to_lds_b128 %0, %1, off offset:48" :: "v"(la), "v"(ga) : "memory");
}

// ---------------------------------------------------------------------------
// Prepass: f32 -> bf16 bulk convert (memory-bound; done once so the GEMMs
// read half the bytes and can stage with pure async copies).
// ---------------------------------------------------------------------------
__global__ __launch_bounds__(256)
void f32_to_bf16_kernel(const float* __restrict__ src, bf16_t* __restrict__ dst, int n) {
  const int i = (blockIdx.x * 256 + threadIdx.x) * 8;
  if (i + 8 <= n) {
    float4 a = *(const float4*)(src + i);
    float4 b = *(const float4*)(src + i + 4);
    v8bf v;
    v[0] = f2bf(a.x); v[1] = f2bf(a.y); v[2] = f2bf(a.z); v[3] = f2bf(a.w);
    v[4] = f2bf(b.x); v[5] = f2bf(b.y); v[6] = f2bf(b.z); v[7] = f2bf(b.w);
    *(v8bf*)(dst + i) = v;
  }
}

// ---------------------------------------------------------------------------
// GEMM with bias: Y[m, n] = sum_k A[m, k] * W[n, k] + bias[n]   (A, W bf16)
// OUT_MODE 0: bf16 to [b*H+h][l][64] (Q, K)
// OUT_MODE 1: bf16 to [b*H+h][64][l] (V transposed, for P*V B-frags)
// OUT_MODE 2: f32  to d_out [m][DIMN] (final projection)
// Block: 256 threads = 8 waves. WG tile 128(M) x 128(N); wave tile 32 x 64.
// K-step 64, double-buffered async global->LDS staging (8 b128 ops/tile).
// ---------------------------------------------------------------------------
template <int OUT_MODE>
__global__ __launch_bounds__(256)
void gemm_bias_kernel(const bf16_t* __restrict__ A,
                      const bf16_t* __restrict__ W,
                      const float* __restrict__ bias,
                      void* __restrict__ Dst) {
  __shared__ bf16_t As[2][128][64];   // 2 x 16 KB
  __shared__ bf16_t Bs[2][128][64];   // 2 x 16 KB

  const int t    = threadIdx.x;
  const int lane = t & 31;
  const int wave = t >> 5;       // 0..7
  const int mw   = wave & 3;     // M subtile pair (32 rows)
  const int nh   = wave >> 2;    // N half (64 cols)
  const int M0   = blockIdx.y * 128;
  const int N0   = blockIdx.x * 128;

  // staging: 128 rows x 64 k; 64 B per thread per array
  const int srow = t >> 1, scol = (t & 1) * 32;

  v8f acc[2][4] = {{{}, {}, {}, {}}, {{}, {}, {}, {}}};

  auto stage = [&](int kt, int buf) {
    async_copy64(A + (size_t)(M0 + srow) * DIMN + kt * 64 + scol, &As[buf][srow][scol]);
    async_copy64(W + (size_t)(N0 + srow) * DIMN + kt * 64 + scol, &Bs[buf][srow][scol]);
  };

  const int NK = DIMN / 64;  // 16
  stage(0, 0);
  for (int it = 0; it < NK; ++it) {
    if (it + 1 < NK) {
      stage(it + 1, (it + 1) & 1);
      asm volatile("s_wait_asynccnt 0x8" ::: "memory");  // tile `it` landed
    } else {
      asm volatile("s_wait_asynccnt 0x0" ::: "memory");
    }
    __syncthreads();

    const int buf = it & 1;
#pragma unroll
    for (int ks = 0; ks < 2; ++ks) {
      v16bf af0 = load_frag(&As[buf][mw * 32][ks * 32],      64, lane);
      v16bf af1 = load_frag(&As[buf][mw * 32 + 16][ks * 32], 64, lane);
#pragma unroll
      for (int tn = 0; tn < 4; ++tn) {
        v16bf bfg = load_frag(&Bs[buf][nh * 64 + tn * 16][ks * 32], 64, lane);
        acc[0][tn] = wmma_bf16(af0, bfg, acc[0][tn]);
        acc[1][tn] = wmma_bf16(af1, bfg, acc[1][tn]);
      }
    }
    __syncthreads();  // all reads done before this buffer is overwritten
  }

  // ---- epilogue: C layout is row = (lane>>4)*8 + r, col = lane&15 ----
  const int col = lane & 15;
  const int rof = (lane >> 4) * 8;
#pragma unroll
  for (int ms = 0; ms < 2; ++ms) {
#pragma unroll
    for (int tn = 0; tn < 4; ++tn) {
      const int n  = N0 + nh * 64 + tn * 16 + col;
      const float bv = bias[n];
#pragma unroll
      for (int r = 0; r < 8; ++r) {
        const int m = M0 + mw * 32 + ms * 16 + rof + r;
        const float y = acc[ms][tn][r] + bv;
        if (OUT_MODE == 2) {
          ((float*)Dst)[(size_t)m * DIMN + n] = y;
        } else {
          const int bi = m >> 11, li = m & (SEQ - 1);
          const int h  = n >> 6,  d  = n & 63;
          bf16_t* o = (bf16_t*)Dst;
          if (OUT_MODE == 0)
            o[((size_t)(bi * HEADS + h) * SEQ + li) * DHEAD + d] = f2bf(y);
          else
            o[((size_t)(bi * HEADS + h) * DHEAD + d) * SEQ + li] = f2bf(y);
        }
      }
    }
  }
}

// ---------------------------------------------------------------------------
// Flash-style attention. Block = 128 threads (4 waves). blockIdx.y = (b*H+h)
// plane, blockIdx.x selects 128 queries; each wave owns two 16-query tiles
// and streams all 2048 keys in tiles of 32. K/V tiles are double-buffered in
// LDS via async global->LDS copies (4 b128 ops/tile), so staging of tile j+1
// overlaps the 16 WMMAs + softmax of tile j.
// ---------------------------------------------------------------------------
__global__ __launch_bounds__(128)
void attention_kernel(const bf16_t* __restrict__ Q,
                      const bf16_t* __restrict__ K,
                      const bf16_t* __restrict__ V,
                      bf16_t* __restrict__ O) {
  __shared__ bf16_t Ks[2][32][64];    // 2 x 4 KB  keys   [j][d]
  __shared__ bf16_t Vs[2][64][32];    // 2 x 4 KB  values [d][j] (V stored transposed)
  __shared__ bf16_t Ps[4][16][32];    // 4 KB      per-wave P tile (C- -> A-layout)

  const int t     = threadIdx.x;
  const int lane  = t & 31;
  const int wave  = t >> 5;
  const int plane = blockIdx.y;                    // b*HEADS + h
  const int l0    = (blockIdx.x * 4 + wave) * 32;  // 32 queries per wave

  const bf16_t* Qp = Q + (size_t)plane * SEQ * DHEAD;
  const bf16_t* Kp = K + (size_t)plane * SEQ * DHEAD;
  const bf16_t* Vp = V + (size_t)plane * DHEAD * SEQ;

  const float SCALE = 0.03125f;  // DIM^-0.5 = 1/32 (full dim, as in reference)

  // Q fragments: 2 q-tiles x (K-dim 64 -> two K=32 frags)
  v16bf qa[2][2];
#pragma unroll
  for (int qt = 0; qt < 2; ++qt) {
    const bf16_t* qb = Qp + (size_t)(l0 + qt * 16) * DHEAD;
    qa[qt][0] = load_frag(qb,      DHEAD, lane);
    qa[qt][1] = load_frag(qb + 32, DHEAD, lane);
  }

  float mrow[2][8], lsum[2][8];
  v8f oacc[2][4] = {{{}, {}, {}, {}}, {{}, {}, {}, {}}};
#pragma unroll
  for (int qt = 0; qt < 2; ++qt)
#pragma unroll
    for (int r = 0; r < 8; ++r) { mrow[qt][r] = -1e30f; lsum[qt][r] = 0.0f; }

  const int krow = t >> 2, kcol = (t & 3) * 16;  // 32 x 64 staging (32B/thread)
  const int vrow = t >> 1, vcol = (t & 1) * 16;  // 64 x 32 staging (32B/thread)

  auto stage = [&](int jt, int buf) {
    async_copy32(Kp + (size_t)(jt * 32 + krow) * DHEAD + kcol, &Ks[buf][krow][kcol]);
    async_copy32(Vp + (size_t)vrow * SEQ + jt * 32 + vcol,     &Vs[buf][vrow][vcol]);
  };

  const int NJ = SEQ / 32;  // 64
  stage(0, 0);
  for (int it = 0; it < NJ; ++it) {
    if (it + 1 < NJ) {
      stage(it + 1, (it + 1) & 1);
      asm volatile("s_wait_asynccnt 0x4" ::: "memory");
    } else {
      asm volatile("s_wait_asynccnt 0x0" ::: "memory");
    }
    __syncthreads();

    const int buf = it & 1;
#pragma unroll
    for (int qt = 0; qt < 2; ++qt) {
      // ---- S = Q K^T : two 16x16 subtiles, K-dim 64 = 2 WMMAs each ----
      v8f s0 = {}, s1 = {};
      s0 = wmma_bf16(qa[qt][0], load_frag(&Ks[buf][0][0],   64, lane), s0);
      s0 = wmma_bf16(qa[qt][1], load_frag(&Ks[buf][0][32],  64, lane), s0);
      s1 = wmma_bf16(qa[qt][0], load_frag(&Ks[buf][16][0],  64, lane), s1);
      s1 = wmma_bf16(qa[qt][1], load_frag(&Ks[buf][16][32], 64, lane), s1);

      // ---- online softmax over the 32 new columns ----
#pragma unroll
      for (int r = 0; r < 8; ++r) {
        float a0 = s0[r] * SCALE;
        float a1 = s1[r] * SCALE;
        float mx = fmaxf(a0, a1);
#pragma unroll
        for (int off = 1; off < 16; off <<= 1)
          mx = fmaxf(mx, __shfl_xor(mx, off, 32));
        const float mnew = fmaxf(mrow[qt][r], mx);
        const float corr = __expf(mrow[qt][r] - mnew);
        mrow[qt][r] = mnew;
        const float p0 = __expf(a0 - mnew);
        const float p1 = __expf(a1 - mnew);
        float rs = p0 + p1;
#pragma unroll
        for (int off = 1; off < 16; off <<= 1)
          rs += __shfl_xor(rs, off, 32);
        lsum[qt][r] = lsum[qt][r] * corr + rs;
#pragma unroll
        for (int dt = 0; dt < 4; ++dt) oacc[qt][dt][r] *= corr;
        // scatter P (C-layout) into this wave's LDS slice for A-layout reload
        const int prow = (lane >> 4) * 8 + r;
        Ps[wave][prow][lane & 15]        = f2bf(p0);
        Ps[wave][prow][16 + (lane & 15)] = f2bf(p1);
      }
      // same-wave LDS RAW: drain DS counter before re-reading Ps
      asm volatile("s_wait_dscnt 0" ::: "memory");

      // ---- O += P * V : one K=32 WMMA per 16-wide d tile ----
      v16bf pa = load_frag(&Ps[wave][0][0], 32, lane);
#pragma unroll
      for (int dt = 0; dt < 4; ++dt) {
        v16bf vb = load_frag(&Vs[buf][dt * 16][0], 32, lane);
        oacc[qt][dt] = wmma_bf16(pa, vb, oacc[qt][dt]);
      }
    }
    __syncthreads();
  }

  // ---- normalize and write attn output as bf16 [b][l][h*64+d] ----
  const int b   = plane >> 4, h = plane & 15;
  const int col = lane & 15, rof = (lane >> 4) * 8;
#pragma unroll
  for (int qt = 0; qt < 2; ++qt) {
#pragma unroll
    for (int dt = 0; dt < 4; ++dt) {
#pragma unroll
      for (int r = 0; r < 8; ++r) {
        const int li = l0 + qt * 16 + rof + r;
        const int d  = dt * 16 + col;
        O[((size_t)(b * SEQ + li)) * DIMN + h * DHEAD + d] =
            f2bf(oacc[qt][dt][r] / lsum[qt][r]);
      }
    }
  }
}

extern "C" void kernel_launch(void* const* d_in, const int* in_sizes, int n_in,
                              void* d_out, int out_size, void* d_ws, size_t ws_size,
                              hipStream_t stream) {
  const float* x  = (const float*)d_in[0];
  const float* wq = (const float*)d_in[1];
  const float* bq = (const float*)d_in[2];
  const float* wk = (const float*)d_in[3];
  const float* bk = (const float*)d_in[4];
  const float* wv = (const float*)d_in[5];
  const float* bv = (const float*)d_in[6];
  const float* wo = (const float*)d_in[7];
  const float* bo = (const float*)d_in[8];

  // workspace layout (bf16): x, 4 weights, Q, K, V^T, attn-out
  const size_t NX = (size_t)MTOT * DIMN;   // 4M
  const size_t NW = (size_t)DIMN * DIMN;   // 1M
  bf16_t* xb  = (bf16_t*)d_ws;
  bf16_t* wqb = xb  + NX;
  bf16_t* wkb = wqb + NW;
  bf16_t* wvb = wkb + NW;
  bf16_t* wob = wvb + NW;
  bf16_t* qws = wob + NW;
  bf16_t* kws = qws + NX;
  bf16_t* vws = kws + NX;
  bf16_t* ows = vws + NX;

  // prepass: bulk f32 -> bf16 (memory-bound, ~50 MB total)
  f32_to_bf16_kernel<<<(int)(NX / (256 * 8)), 256, 0, stream>>>(x,  xb,  (int)NX);
  f32_to_bf16_kernel<<<(int)(NW / (256 * 8)), 256, 0, stream>>>(wq, wqb, (int)NW);
  f32_to_bf16_kernel<<<(int)(NW / (256 * 8)), 256, 0, stream>>>(wk, wkb, (int)NW);
  f32_to_bf16_kernel<<<(int)(NW / (256 * 8)), 256, 0, stream>>>(wv, wvb, (int)NW);
  f32_to_bf16_kernel<<<(int)(NW / (256 * 8)), 256, 0, stream>>>(wo, wob, (int)NW);

  const dim3 gproj(DIMN / 128, MTOT / 128);  // 8 x 32
  gemm_bias_kernel<0><<<gproj, 256, 0, stream>>>(xb, wqb, bq, qws);
  gemm_bias_kernel<0><<<gproj, 256, 0, stream>>>(xb, wkb, bk, kws);
  gemm_bias_kernel<1><<<gproj, 256, 0, stream>>>(xb, wvb, bv, vws);

  const dim3 gattn(SEQ / 128, BATCH * HEADS);  // 16 x 32
  attention_kernel<<<gattn, 128, 0, stream>>>(qws, kws, vws, ows);

  gemm_bias_kernel<2><<<gproj, 256, 0, stream>>>(ows, wob, bo, d_out);
}
